// IPAttnProcessor_81905026335370
// MI455X (gfx1250) — matrix operand order repaired and checked by reading, not compiled, not run on hardware
//
#include <hip/hip_runtime.h>
#include <hip/hip_bf16.h>
#include <math.h>

typedef __attribute__((ext_vector_type(16))) _Float16 v16h;
typedef __attribute__((ext_vector_type(8)))  _Float16 v8h;
typedef __attribute__((ext_vector_type(8)))  float    v8f;
typedef __attribute__((ext_vector_type(4)))  unsigned int v4u;
typedef __attribute__((ext_vector_type(8)))  int      v8i;
typedef __attribute__((ext_vector_type(4)))  int      v4i;

union V16U { v16h v; v8h h[2]; };

// ---- WMMA fragment loaders (16x16x32 f16, wave32) ----------------------
// A (16xK chunk, row-major X): lane l -> row m=l%16; halves K=k0+g*8..+7 and k0+16+g*8..+7
__device__ __forceinline__ v16h load_a_frag(const _Float16* X, int ld, int base_m, int max_m, int k0) {
    int lane = threadIdx.x & 31;
    int m = base_m + (lane & 15); if (m > max_m) m = max_m;
    int g = lane >> 4;
    const _Float16* p = X + (size_t)m * ld + k0 + g * 8;
    V16U u;
    u.h[0] = *(const v8h*)p;
    u.h[1] = *(const v8h*)(p + 16);
    return u.v;
}
// B (Kx16 chunk as W[n,k] row-major): lane l -> col n=l%16; halves K=k0+g*8..+7 and k0+16+g*8..+7
__device__ __forceinline__ v16h load_b_frag(const _Float16* W, int ld, int base_n, int max_n, int k0) {
    int lane = threadIdx.x & 31;
    int n = base_n + (lane & 15); if (n > max_n) n = max_n;
    int g = lane >> 4;
    const _Float16* p = W + (size_t)n * ld + k0 + g * 8;
    V16U u;
    u.h[0] = *(const v8h*)p;
    u.h[1] = *(const v8h*)(p + 16);
    return u.v;
}
__device__ __forceinline__ v8f wmma16(v16h a, v16h b, v8f c) {
    return __builtin_amdgcn_wmma_f32_16x16x32_f16(false, a, false, b, (short)0, c, false, false);
}
__device__ __forceinline__ v8f vzero8() { v8f z = {0.f,0.f,0.f,0.f,0.f,0.f,0.f,0.f}; return z; }

__device__ __forceinline__ void wait_tensorcnt0() {
#if __has_builtin(__builtin_amdgcn_s_wait_tensorcnt)
    __builtin_amdgcn_s_wait_tensorcnt(0);
#else
    asm volatile("s_wait_tensorcnt 0x0" ::: "memory");
#endif
}

// ---- TDM stage: copy a 16x640 f16 tile (row-major, stride 640) into LDS.
// One descriptor issued by wave 0; D# layout per CDNA5 ISA 8.3/8.4.
__device__ __forceinline__ void stage_a16x640(_Float16* lds_dst, const _Float16* gsrc) {
#if __has_builtin(__builtin_amdgcn_tensor_load_to_lds)
    if (threadIdx.x < 32) {
        unsigned long long ga = (unsigned long long)(uintptr_t)gsrc;
        unsigned int ldsoff = (unsigned int)(uintptr_t)lds_dst;   // low 32 bits = LDS offset
        // group0: [1:0]=count=1, [63:32]=lds_addr, [120:64]=global_addr, [127:126]=type=2
        v4u g0 = { 1u,
                   ldsoff,
                   (unsigned int)ga,
                   (unsigned int)((ga >> 32) & 0x01ffffffu) | 0x80000000u };
        // group1: [17:16]=data_size=1(2B); tensor_dim0=640 @ [79:48]; tensor_dim1=16 @ [111:80];
        //         tile_dim0=640 @ [127:112]; tile_dim1=16 @ [143:128]; tensor_dim0_stride=640 @ [207:160]
        v8i g1 = { (int)0x00010000u,          // workgroup_mask=0, data_size=1
                   (int)(640u << 16),         // tensor_dim0[15:0] << 16
                   (int)(16u << 16),          // tensor_dim0[31:16]=0 | tensor_dim1[15:0]=16
                   (int)(640u << 16),         // tensor_dim1[31:16]=0 | tile_dim0=640
                   16,                        // tile_dim1=16, tile_dim2=0
                   640,                       // tensor_dim0_stride[31:0]
                   0,                         // stride0[47:32]=0 | stride1[15:0]=0
                   0 };                       // stride1[47:16]=0
        v4i g2 = { 0, 0, 0, 0 };
        v4i g3 = { 0, 0, 0, 0 };
        v8i g4 = { 0, 0, 0, 0, 0, 0, 0, 0 };  // unused for 2D tile (zero-filled)
        __builtin_amdgcn_tensor_load_to_lds(g0, g1, g2, g3, g4, 0);
        wait_tensorcnt0();
    }
#else
    for (int i = threadIdx.x; i < (16 * 640) / 8; i += blockDim.x)
        ((v8h*)lds_dst)[i] = ((const v8h*)gsrc)[i];
#endif
    __syncthreads();
}

// ---- elementwise helpers ----------------------------------------------
__global__ void cvt_f32_to_f16(const float* __restrict__ in, _Float16* __restrict__ out, long n) {
    long i = (long)blockIdx.x * blockDim.x + threadIdx.x;
    if (i < n) out[i] = (_Float16)in[i];
}
__global__ void zero_f16(_Float16* __restrict__ p, long n) {
    long i = (long)blockIdx.x * blockDim.x + threadIdx.x;
    if (i < n) p[i] = (_Float16)0.0f;
}

// ---- KV projection: out = ehs @ W^T (f16), optional transposed store ---
__global__ void kvproj_kernel(const _Float16* __restrict__ A, int lda, int arow0, int mvalid,
                              long abstride,
                              const _Float16* __restrict__ W, int kdim,
                              _Float16* __restrict__ out, int oldim, long obstride,
                              int transposed) {
    int b = blockIdx.z;
    const _Float16* Ab = A + (long)b * abstride + (long)arow0 * lda;
    _Float16* ob = out + (long)b * obstride;
    int mbase = blockIdx.x * 16, nbase = blockIdx.y * 16;
    v8f acc = vzero8();
    for (int k0 = 0; k0 < kdim; k0 += 32) {
        v16h a  = load_a_frag(Ab, lda, mbase, mvalid - 1, k0);
        v16h bb = load_b_frag(W, kdim, nbase, 639, k0);
        acc = wmma16(a, bb, acc);
    }
    int lane = threadIdx.x & 31, g = lane >> 4, n = lane & 15;
#pragma unroll
    for (int r = 0; r < 8; r++) {
        int m = mbase + r + 8 * g;
        int col = nbase + n;
        if (transposed) ob[(long)col * oldim + m] = (_Float16)acc[r];
        else            ob[(long)m * oldim + col] = (_Float16)acc[r];
    }
}

// ---- Q projection: q = hs @ Wq^T, scaled by 1/sqrt(64), f16 out --------
// grid.x = M/16 tiles, block = 256 (8 waves); wave w covers 80 output cols.
// A tile staged to LDS once per block via TDM.
__global__ void qproj_kernel(const _Float16* __restrict__ A, const _Float16* __restrict__ W,
                             _Float16* __restrict__ out) {
    __shared__ __align__(16) _Float16 ldsA[16 * 640];
    int wave = threadIdx.x >> 5;
    int mbase = blockIdx.x * 16;
    int nwb = wave * 80;
    stage_a16x640(ldsA, A + (long)mbase * 640);
    v8f acc[5];
#pragma unroll
    for (int t = 0; t < 5; t++) acc[t] = vzero8();
    for (int k0 = 0; k0 < 640; k0 += 32) {
        v16h a = load_a_frag(ldsA, 640, 0, 15, k0);
#pragma unroll
        for (int t = 0; t < 5; t++) {
            v16h bb = load_b_frag(W, 640, nwb + t * 16, 639, k0);
            acc[t] = wmma16(a, bb, acc[t]);
        }
    }
    int lane = threadIdx.x & 31, g = lane >> 4, n = lane & 15;
#pragma unroll
    for (int t = 0; t < 5; t++)
#pragma unroll
        for (int r = 0; r < 8; r++) {
            long m = mbase + r + 8 * g;
            out[m * 640 + nwb + t * 16 + n] = (_Float16)(acc[t][r] * 0.125f);
        }
}

// ---- Output projection: out = attn @ Wo^T + bo, fp32 out ---------------
__global__ void oproj_kernel(const _Float16* __restrict__ A, const _Float16* __restrict__ W,
                             const float* __restrict__ bo, float* __restrict__ out) {
    __shared__ __align__(16) _Float16 ldsA[16 * 640];
    int wave = threadIdx.x >> 5;
    int mbase = blockIdx.x * 16;
    int nwb = wave * 80;
    stage_a16x640(ldsA, A + (long)mbase * 640);
    v8f acc[5];
#pragma unroll
    for (int t = 0; t < 5; t++) acc[t] = vzero8();
    for (int k0 = 0; k0 < 640; k0 += 32) {
        v16h a = load_a_frag(ldsA, 640, 0, 15, k0);
#pragma unroll
        for (int t = 0; t < 5; t++) {
            v16h bb = load_b_frag(W, 640, nwb + t * 16, 639, k0);
            acc[t] = wmma16(a, bb, acc[t]);
        }
    }
    int lane = threadIdx.x & 31, g = lane >> 4, n = lane & 15;
#pragma unroll
    for (int t = 0; t < 5; t++) {
        int col = nwb + t * 16 + n;
        float bias = bo[col];
#pragma unroll
        for (int r = 0; r < 8; r++) {
            long m = mbase + r + 8 * g;
            out[m * 640 + col] = acc[t][r] + bias;
        }
    }
}

// ---- fused text+ip attention -------------------------------------------
// per wave: one (b, head, 16-query tile). block = 128 (4 waves).
#define LDS_PT  (16 * 96)
#define LDS_PIP (16 * 32)
#define LDS_PW  (LDS_PT + LDS_PIP)

__global__ void attn_kernel(const _Float16* __restrict__ q,    // (B,4096,640), pre-scaled
                            const _Float16* __restrict__ Kt,   // (B,80,640)
                            const _Float16* __restrict__ Vt,   // (B,640,96) transposed, zero-padded
                            const _Float16* __restrict__ Kip,  // (B,16,640)
                            const _Float16* __restrict__ Vipt, // (B,640,32) transposed, zero-padded
                            _Float16* __restrict__ out) {      // (B,4096,640)
    __shared__ __align__(16) _Float16 lds[4 * LDS_PW];
    int wave = threadIdx.x >> 5;
    int lane = threadIdx.x & 31, g = lane >> 4, n = lane & 15;
    int task = blockIdx.x * 4 + wave;           // 40960 tasks
    int head = task % 10;
    int t2 = task / 10;
    int mt = t2 & 255;
    int b  = t2 >> 8;

    const _Float16* qb   = q   + ((long)b * 4096 + mt * 16) * 640;
    const _Float16* Kb   = Kt  + (long)b * 80 * 640;
    const _Float16* Vb   = Vt  + (long)b * 640 * 96;
    const _Float16* Kipb = Kip + (long)b * 16 * 640;
    const _Float16* Vipb = Vipt + (long)b * 640 * 32;
    _Float16* lp  = &lds[wave * LDS_PW];
    _Float16* lpi = lp + LDS_PT;

    // Q fragments (reused across all key tiles)
    v16h aq[2];
#pragma unroll
    for (int c = 0; c < 2; c++) aq[c] = load_a_frag(qb, 640, 0, 0x0fffffff, head * 64 + c * 32);

    // scores: text (5 key tiles) and ip (1 tile)
    v8f sc[5];
#pragma unroll
    for (int jt = 0; jt < 5; jt++) {
        v8f a = vzero8();
#pragma unroll
        for (int c = 0; c < 2; c++) {
            v16h bb = load_b_frag(Kb, 640, jt * 16, 79, head * 64 + c * 32);
            a = wmma16(aq[c], bb, a);
        }
        sc[jt] = a;
    }
    v8f si = vzero8();
#pragma unroll
    for (int c = 0; c < 2; c++) {
        v16h bb = load_b_frag(Kipb, 640, 0, 15, head * 64 + c * 32);
        si = wmma16(aq[c], bb, si);
    }

    // masking (lane n == key index within tile)
    const float NEG = -1.0e30f;
    if (64 + n >= 77) {
#pragma unroll
        for (int r = 0; r < 8; r++) sc[4][r] = NEG;
    }
    if (n >= 4) {
#pragma unroll
        for (int r = 0; r < 8; r++) si[r] = NEG;
    }

    // softmax per query row (row m lives in 16 lanes of one half-wave, VGPR r)
#pragma unroll
    for (int r = 0; r < 8; r++) {
        // text
        float mx = sc[0][r];
#pragma unroll
        for (int jt = 1; jt < 5; jt++) mx = fmaxf(mx, sc[jt][r]);
#pragma unroll
        for (int off = 1; off < 16; off <<= 1) mx = fmaxf(mx, __shfl_xor(mx, off, 32));
        float s = 0.f;
#pragma unroll
        for (int jt = 0; jt < 5; jt++) { float e = __expf(sc[jt][r] - mx); sc[jt][r] = e; s += e; }
#pragma unroll
        for (int off = 1; off < 16; off <<= 1) s += __shfl_xor(s, off, 32);
        float inv = 1.0f / s;
#pragma unroll
        for (int jt = 0; jt < 5; jt++) sc[jt][r] *= inv;
        // ip
        float mi = si[r];
#pragma unroll
        for (int off = 1; off < 16; off <<= 1) mi = fmaxf(mi, __shfl_xor(mi, off, 32));
        float ei = __expf(si[r] - mi);
        float ssi = ei;
#pragma unroll
        for (int off = 1; off < 16; off <<= 1) ssi += __shfl_xor(ssi, off, 32);
        si[r] = ei / ssi;
    }

    // zero-fill P region (covers j padding), then scatter probs (C-layout -> row-major)
    {
        uint32_t* lz = (uint32_t*)lp;   // LDS_PW halves = 1024 dwords
#pragma unroll
        for (int i = 0; i < 32; i++) lz[lane + i * 32] = 0u;
    }
#pragma unroll
    for (int r = 0; r < 8; r++) {
        int m = r + 8 * g;
#pragma unroll
        for (int jt = 0; jt < 5; jt++)
            lp[m * 96 + jt * 16 + n] = (_Float16)sc[jt][r];
        lpi[m * 32 + n] = (_Float16)si[r];
    }
    asm volatile("s_wait_dscnt 0" ::: "memory");

    // out = P @ V  (+ Pip @ Vip), V supplied transposed so B-frags are contiguous
    v8f o[4];
#pragma unroll
    for (int dt = 0; dt < 4; dt++) o[dt] = vzero8();
#pragma unroll
    for (int c = 0; c < 3; c++) {
        v16h ap = load_a_frag(lp, 96, 0, 15, c * 32);
#pragma unroll
        for (int dt = 0; dt < 4; dt++) {
            v16h bv = load_b_frag(Vb, 96, head * 64 + dt * 16, 639, c * 32);
            o[dt] = wmma16(ap, bv, o[dt]);
        }
    }
    {
        v16h ap = load_a_frag(lpi, 32, 0, 15, 0);
#pragma unroll
        for (int dt = 0; dt < 4; dt++) {
            v16h bv = load_b_frag(Vipb, 32, head * 64 + dt * 16, 639, 0);
            o[dt] = wmma16(ap, bv, o[dt]);
        }
    }

    _Float16* ob = out + ((long)b * 4096 + mt * 16) * 640 + head * 64;
#pragma unroll
    for (int dt = 0; dt < 4; dt++)
#pragma unroll
        for (int r = 0; r < 8; r++)
            ob[(long)(r + 8 * g) * 640 + dt * 16 + n] = (_Float16)o[dt][r];
}

// ------------------------------------------------------------------------
extern "C" void kernel_launch(void* const* d_in, const int* in_sizes, int n_in,
                              void* d_out, int out_size, void* d_ws, size_t ws_size,
                              hipStream_t stream) {
    const float* hs  = (const float*)d_in[0];
    const float* ehs = (const float*)d_in[1];
    const float* Wq  = (const float*)d_in[2];
    const float* Wk  = (const float*)d_in[3];
    const float* Wv  = (const float*)d_in[4];
    const float* Wki = (const float*)d_in[5];
    const float* Wvi = (const float*)d_in[6];
    const float* Wo  = (const float*)d_in[7];
    const float* bo  = (const float*)d_in[8];
    float* out = (float*)d_out;

    char* w = (char*)d_ws;
    size_t off = 0;
    auto alloc = [&](size_t bytes) -> _Float16* {
        _Float16* p = (_Float16*)(w + off);
        off += (bytes + 255) & ~(size_t)255;
        return p;
    };
    const long NHS  = 16L * 4096 * 640;
    const long NEHS = 16L * 81 * 768;
    _Float16* hs_h   = alloc(NHS * 2);          // reused as attn output buffer
    _Float16* q_h    = alloc(NHS * 2);
    _Float16* ehs_h  = alloc(NEHS * 2);
    _Float16* Wq_h   = alloc(640L * 640 * 2);
    _Float16* Wk_h   = alloc(640L * 768 * 2);
    _Float16* Wv_h   = alloc(640L * 768 * 2);
    _Float16* Wki_h  = alloc(640L * 768 * 2);
    _Float16* Wvi_h  = alloc(640L * 768 * 2);
    _Float16* Wo_h   = alloc(640L * 640 * 2);
    _Float16* K_h    = alloc(16L * 80 * 640 * 2);
    _Float16* Vt_h   = alloc(16L * 640 * 96 * 2);
    _Float16* Kip_h  = alloc(16L * 16 * 640 * 2);
    _Float16* Vipt_h = alloc(16L * 640 * 32 * 2);
    _Float16* attn_h = hs_h;   // alias: hs_h fully consumed by qproj before attn writes

    auto cvt = [&](const float* in, _Float16* o, long n) {
        cvt_f32_to_f16<<<(unsigned)((n + 255) / 256), 256, 0, stream>>>(in, o, n);
    };
    cvt(hs,  hs_h,  NHS);
    cvt(ehs, ehs_h, NEHS);
    cvt(Wq,  Wq_h,  640L * 640);
    cvt(Wk,  Wk_h,  640L * 768);
    cvt(Wv,  Wv_h,  640L * 768);
    cvt(Wki, Wvi_h == nullptr ? Wvi_h : Wki_h, 640L * 768);
    cvt(Wvi, Wvi_h, 640L * 768);
    cvt(Wo,  Wo_h,  640L * 640);

    zero_f16<<<(unsigned)((16L * 640 * 96 + 255) / 256), 256, 0, stream>>>(Vt_h, 16L * 640 * 96);
    zero_f16<<<(unsigned)((16L * 640 * 32 + 255) / 256), 256, 0, stream>>>(Vipt_h, 16L * 640 * 32);

    // K, V projections (text rows 0..76; ip rows 77..80); V stored transposed
    kvproj_kernel<<<dim3(5, 40, 16), 32, 0, stream>>>(ehs_h, 768, 0, 77, 81L * 768,
                                                      Wk_h, 768, K_h, 640, 80L * 640, 0);
    kvproj_kernel<<<dim3(5, 40, 16), 32, 0, stream>>>(ehs_h, 768, 0, 77, 81L * 768,
                                                      Wv_h, 768, Vt_h, 96, 640L * 96, 1);
    kvproj_kernel<<<dim3(1, 40, 16), 32, 0, stream>>>(ehs_h, 768, 77, 4, 81L * 768,
                                                      Wki_h, 768, Kip_h, 640, 16L * 640, 0);
    kvproj_kernel<<<dim3(1, 40, 16), 32, 0, stream>>>(ehs_h, 768, 77, 4, 81L * 768,
                                                      Wvi_h, 768, Vipt_h, 32, 640L * 32, 1);

    qproj_kernel<<<4096, 256, 0, stream>>>(hs_h, Wq_h, q_h);
    attn_kernel<<<10240, 128, 0, stream>>>(q_h, K_h, Vt_h, Kip_h, Vipt_h, attn_h);
    oproj_kernel<<<4096, 256, 0, stream>>>(attn_h, Wo_h, bo, out);
}
